// MinEuclideanDistBlock_79929341378724
// MI455X (gfx1250) — matrix operand order repaired and checked by reading, not compiled, not run on hardware
//
#include <hip/hip_runtime.h>
#include <hip/hip_bf16.h>

// ---------------------------------------------------------------------------
// Shapelet network for MI455X (gfx1250, wave32, WMMA + async-to-LDS).
// Activations are stored L-major bf16: act[n][l][k]. A conv input tile
// [col][ki] is then a contiguous global region == the column-major LDS tile,
// so tiles are staged with GLOBAL_LOAD_ASYNC_TO_LDS_B128 (ASYNCcnt) and all
// WMMA B fragments are two ds_load_b128. Repacked weights give A fragments
// as two global_load_b128, each reused by 4 WMMAs (4 l-subtiles per wave).
// Epilogues pack 8 bf16 results into one global_store_b128 per subtile.
// Transcendentals use native v_exp_f32 / v_sqrt_f32 / v_rcp_f32.
// ---------------------------------------------------------------------------

#define B_DIM   32
#define C_DIM   3
#define L_DIM   2048
#define S_DIM   64
#define K_DIM   256
#define NS      (B_DIM * C_DIM)   // 96 samples
#define DK_DIM  128

typedef __bf16 bf16_t;
typedef bf16_t v16bf __attribute__((ext_vector_type(16)));
typedef float  v8f   __attribute__((ext_vector_type(8)));

union ABPack { unsigned int u[8]; v16bf v; };

__device__ __forceinline__ unsigned short f2bf(float f) {
  unsigned int u = __float_as_uint(f);
  u += 0x7FFFu + ((u >> 16) & 1u);          // round-to-nearest-even
  return (unsigned short)(u >> 16);
}
__device__ __forceinline__ float bf2f(unsigned short h) {
  return __uint_as_float(((unsigned int)h) << 16);
}
__device__ __forceinline__ unsigned int pk2(float a, float b) {
  return (unsigned int)f2bf(a) | ((unsigned int)f2bf(b) << 16);
}
// tanh-form GELU: one v_exp_f32 + one v_rcp_f32 (no IEEE div expansion)
__device__ __forceinline__ float gelu_fast(float x) {
  float g = 0.7978845608028654f * (x + 0.044715f * x * x * x);
  float e = __expf(-2.f * g);
  float t = (1.f - e) * __builtin_amdgcn_rcpf(1.f + e);   // tanh(g)
  return 0.5f * x * (1.f + t);
}

// D(16x16,f32) += A(16x32,bf16) x B(32x16,bf16); operands arrive as 4x uint4.
__device__ __forceinline__ v8f wmma_uv(uint4 alo, uint4 ahi, uint4 blo, uint4 bhi, v8f acc) {
  ABPack a, b;
  a.u[0] = alo.x; a.u[1] = alo.y; a.u[2] = alo.z; a.u[3] = alo.w;
  a.u[4] = ahi.x; a.u[5] = ahi.y; a.u[6] = ahi.z; a.u[7] = ahi.w;
  b.u[0] = blo.x; b.u[1] = blo.y; b.u[2] = blo.z; b.u[3] = blo.w;
  b.u[4] = bhi.x; b.u[5] = bhi.y; b.u[6] = bhi.z; b.u[7] = bhi.w;
  return __builtin_amdgcn_wmma_f32_16x16x32_bf16(
      false, a.v, false, b.v, (short)0, acc, false, false);
}

// async copy of one 16B chunk, global -> LDS (tracked by ASYNCcnt)
__device__ __forceinline__ void async_b128(unsigned int lds_off, const void* g) {
  asm volatile("global_load_async_to_lds_b128 %0, %1, off"
               :: "v"(lds_off), "v"((unsigned long long)(size_t)g) : "memory");
}
__device__ __forceinline__ void wait_async0() {
  asm volatile("s_wait_asynccnt 0x0" ::: "memory");
}

// ---------------------------------------------------------------------------
// init: shapelet squared norms + dmin = +inf
// ---------------------------------------------------------------------------
__global__ void k_init(const float* __restrict__ shp, float* __restrict__ s2,
                       float* __restrict__ dmin) {
  int t = blockIdx.x * blockDim.x + threadIdx.x;
  if (t < NS * K_DIM) dmin[t] = __uint_as_float(0x7F800000u);
  if (t < C_DIM * K_DIM) {
    const float* p = shp + t * S_DIM;
    float s = 0.f;
    for (int i = 0; i < S_DIM; ++i) s += p[i] * p[i];
    s2[t] = s;
  }
}

// shapelets f32 -> bf16 (same [c][k][s] layout; s contiguous => b128 A frags)
__global__ void k_scvt(const float* __restrict__ shp, unsigned short* __restrict__ out) {
  int t = blockIdx.x * blockDim.x + threadIdx.x;
  if (t < C_DIM * K_DIM * S_DIM) out[t] = f2bf(shp[t]);
}

// window squared norms: x2[bc][l] = sum_s xp[l+s-31]^2
__global__ void k_x2(const float* __restrict__ x, float* __restrict__ x2) {
  int t = blockIdx.x * blockDim.x + threadIdx.x;
  if (t >= NS * L_DIM) return;
  int bc = t / L_DIM, l = t % L_DIM;
  const float* xp = x + (size_t)bc * L_DIM;
  float s = 0.f;
  for (int si = 0; si < S_DIM; ++si) {
    int xi = l + si - 31;
    float v = (xi >= 0 && xi < L_DIM) ? xp[xi] : 0.f;
    s += v * v;
  }
  x2[t] = s;
}

// weight repack (K,K,T) f32 -> [t][ko][ki] bf16 (ki contiguous => b128 A frags)
__global__ void k_wcvt(const float* __restrict__ w, unsigned short* __restrict__ out, int T) {
  int e = blockIdx.x * blockDim.x + threadIdx.x;
  int total = T * K_DIM * K_DIM;
  if (e >= total) return;
  int t = e / (K_DIM * K_DIM);
  int r = e % (K_DIM * K_DIM);
  int ko = r / K_DIM, ki = r % K_DIM;
  out[e] = f2bf(w[(ko * K_DIM + ki) * T + t]);
}

// ---------------------------------------------------------------------------
// Distance: block = one (b,c) x 16-wide L tile. A = shapelets (M=k rows),
// B = window tile in LDS (N=l cols, ki=s contiguous). D rows are consecutive
// k -> packed b128 stores into ybf[n][l][k]. min_L(d) via half-wave shfl
// reduction + int-bit atomicMin (d >= 0).
// ---------------------------------------------------------------------------
__global__ void __launch_bounds__(256) k_dist(
    const float* __restrict__ x, const unsigned short* __restrict__ shpbf,
    const float* __restrict__ x2, const float* __restrict__ s2,
    unsigned short* __restrict__ ybf, float* __restrict__ dmin) {
  __shared__ __align__(16) unsigned short xwin[16 * 64];  // [l-row][s]
  __shared__ float x2row[16];
  int n  = blockIdx.x >> 7;
  int l0 = (blockIdx.x & 127) << 4;
  int c  = n % C_DIM;
  int tid = threadIdx.x;
  const float* xp = x + (size_t)n * L_DIM;

  for (int e = tid; e < 16 * 64; e += 256) {
    int m = e >> 6, si = e & 63;
    int xi = l0 + m + si - 31;
    float v = (xi >= 0 && xi < L_DIM) ? xp[xi] : 0.f;
    xwin[e] = f2bf(v);
  }
  if (tid < 16) x2row[tid] = x2[(size_t)n * L_DIM + l0 + tid];
  __syncthreads();

  int wave = tid >> 5, lane = tid & 31, h = lane >> 4, q = lane & 15;
  const float* s2c = s2 + c * K_DIM;
  const unsigned short* bcol = xwin + q * 64;    // B column: l = l0+q
  float x2v = x2row[q];

  for (int i = 0; i < 2; ++i) {                  // 8 waves x 2 = 16 k-tiles
    int k0 = (wave * 2 + i) << 4;
    const unsigned short* wrow = shpbf + ((size_t)c * K_DIM + k0 + q) * S_DIM;
    v8f acc = {0.f, 0.f, 0.f, 0.f, 0.f, 0.f, 0.f, 0.f};
#pragma unroll
    for (int kk = 0; kk < S_DIM; kk += 32) {
      uint4 alo = *(const uint4*)(wrow + kk + h * 16);
      uint4 ahi = *(const uint4*)(wrow + kk + h * 16 + 8);
      uint4 blo = *(const uint4*)(bcol + kk + h * 16);
      uint4 bhi = *(const uint4*)(bcol + kk + h * 16 + 8);
      acc = wmma_uv(alo, ahi, blo, bhi, acc);
    }
    float dv[8];
#pragma unroll
    for (int r = 0; r < 8; ++r) {                // D: M(k) = r + 8h, N(l) = q
      float dd = x2v + s2c[k0 + 8 * h + r] - 2.f * acc[r];
      dv[r] = __builtin_amdgcn_sqrtf(fmaxf(dd, 0.f));
    }
    uint4 pk;
    pk.x = pk2(gelu_fast(dv[0]), gelu_fast(dv[1]));
    pk.y = pk2(gelu_fast(dv[2]), gelu_fast(dv[3]));
    pk.z = pk2(gelu_fast(dv[4]), gelu_fast(dv[5]));
    pk.w = pk2(gelu_fast(dv[6]), gelu_fast(dv[7]));
    *(uint4*)(ybf + ((size_t)n * L_DIM + l0 + q) * K_DIM + k0 + 8 * h) = pk;
#pragma unroll
    for (int r = 0; r < 8; ++r) {                // min over the 16 l-lanes
      float m = dv[r];
      m = fminf(m, __shfl_xor(m, 1, 32));
      m = fminf(m, __shfl_xor(m, 2, 32));
      m = fminf(m, __shfl_xor(m, 4, 32));
      m = fminf(m, __shfl_xor(m, 8, 32));
      if (q == 0)
        atomicMin((int*)dmin + n * K_DIM + k0 + 8 * h + r, __float_as_int(m));
    }
  }
}

// K=256 accumulation, A fragments reused across 4 B columns (l-subtiles).
__device__ __forceinline__ void gemm256x4(const unsigned short* __restrict__ wrow,
                                          const unsigned short* bcol0, int h,
                                          v8f acc[4]) {
#pragma unroll
  for (int kk = 0; kk < K_DIM; kk += 32) {
    uint4 alo = *(const uint4*)(wrow + kk + h * 16);
    uint4 ahi = *(const uint4*)(wrow + kk + h * 16 + 8);
    const unsigned short* bc = bcol0 + kk + h * 16;
#pragma unroll
    for (int j = 0; j < 4; ++j) {
      uint4 blo = *(const uint4*)(bc);
      uint4 bhi = *(const uint4*)(bc + 8);
      acc[j] = wmma_uv(alo, ahi, blo, bhi, acc[j]);
      bc += 16 * K_DIM;                          // next l-subtile column
    }
  }
}

// zero-pad edge columns + async-copy the valid contiguous span into LDS
__device__ __forceinline__ void stage_tile(unsigned short* tile,
                                           const unsigned short* act_nl0,  // &act[n][l0-P][0]
                                           int cs, int ce, int tid) {
  for (int col = 0; col < cs; ++col) tile[col * K_DIM + tid] = 0;
  // (ce is the last valid col; cols after it are zeroed by caller's loop bound)
  unsigned int lbase = (unsigned int)(size_t)(tile + cs * K_DIM);
  const char* gsrc = (const char*)(act_nl0 + (size_t)cs * K_DIM);
  int nchunk = (ce - cs + 1) * 32;               // 16B chunks (512B per col)
  for (int ch = tid; ch < nchunk; ch += 256)
    async_b128(lbase + ch * 16, gsrc + (size_t)ch * 16);
}

// ---------------------------------------------------------------------------
// Conv1d (K->K, T taps, 'same'): block = (sample, 64-wide L tile). LDS tile
// column-major [col][ki] staged via global_load_async_to_lds_b128. Each wave:
// 2 ko-tiles x 4 l-subtiles; A fragments loaded once per (t,kk); outputs
// packed into one b128 store per subtile (act layout [n][l][k]).
// ---------------------------------------------------------------------------
template <int T, bool RELU>
__global__ void __launch_bounds__(256) k_conv(
    const unsigned short* __restrict__ in, const unsigned short* __restrict__ w,
    const float* __restrict__ bias, unsigned short* __restrict__ out) {
  constexpr int P  = (T - 1) / 2;
  constexpr int LT = 64;
  constexpr int WC = LT + T - 1;
  __shared__ __align__(16) unsigned short tile[WC * K_DIM];  // [col][ki]
  int n  = blockIdx.x >> 5;                      // 32 l-tiles per sample
  int l0 = (blockIdx.x & 31) * LT;
  int tid = threadIdx.x;

  int cs = (l0 - P < 0) ? (P - l0) : 0;
  int ce = (l0 + LT - 1 + P >= L_DIM) ? (L_DIM - 1 + P - l0) : (WC - 1);
  for (int col = ce + 1; col < WC; ++col) tile[col * K_DIM + tid] = 0;
  stage_tile(tile, in + ((size_t)n * L_DIM + l0 - P) * K_DIM, cs, ce, tid);
  wait_async0();
  __syncthreads();

  int wave = tid >> 5, lane = tid & 31, h = lane >> 4, q = lane & 15;
  for (int i = 0; i < 2; ++i) {
    int ko0 = (wave * 2 + i) << 4;
    v8f acc[4];
#pragma unroll
    for (int j = 0; j < 4; ++j) acc[j] = (v8f){0.f, 0.f, 0.f, 0.f, 0.f, 0.f, 0.f, 0.f};
    for (int t = 0; t < T; ++t)
      gemm256x4(w + (size_t)t * K_DIM * K_DIM + (ko0 + q) * K_DIM,
                tile + (q + t) * K_DIM, h, acc);
    float bs[8];
#pragma unroll
    for (int r = 0; r < 8; ++r) bs[r] = bias[ko0 + 8 * h + r];
#pragma unroll
    for (int j = 0; j < 4; ++j) {
      float v[8];
#pragma unroll
      for (int r = 0; r < 8; ++r) {
        v[r] = acc[j][r] + bs[r];
        if (RELU) v[r] = fmaxf(v[r], 0.f);
      }
      uint4 pk;
      pk.x = pk2(v[0], v[1]); pk.y = pk2(v[2], v[3]);
      pk.z = pk2(v[4], v[5]); pk.w = pk2(v[6], v[7]);
      *(uint4*)(out + ((size_t)n * L_DIM + l0 + 16 * j + q) * K_DIM + ko0 + 8 * h) = pk;
    }
  }
}

// conv3(h2)+b3 + conv1x1(y)+bsc, then ReLU (64-wide L tile)
__global__ void __launch_bounds__(256) k_conv3skip(
    const unsigned short* __restrict__ h2, const unsigned short* __restrict__ y,
    const unsigned short* __restrict__ w3, const unsigned short* __restrict__ wsc,
    const float* __restrict__ b3, const float* __restrict__ bsc,
    unsigned short* __restrict__ out) {
  __shared__ __align__(16) unsigned short tA[66 * K_DIM];   // [col][ki], P=1
  __shared__ __align__(16) unsigned short tY[64 * K_DIM];   // [col][ki]
  int n  = blockIdx.x >> 5;
  int l0 = (blockIdx.x & 31) * 64;
  int tid = threadIdx.x;

  int cs = (l0 == 0) ? 1 : 0;
  int ce = (l0 + 64 >= L_DIM) ? 64 : 65;
  for (int col = ce + 1; col < 66; ++col) tA[col * K_DIM + tid] = 0;
  stage_tile(tA, h2 + ((size_t)n * L_DIM + l0 - 1) * K_DIM, cs, ce, tid);
  {
    unsigned int lbase = (unsigned int)(size_t)tY;
    const char* gsrc = (const char*)(y + ((size_t)n * L_DIM + l0) * K_DIM);
    for (int ch = tid; ch < 64 * 32; ch += 256)
      async_b128(lbase + ch * 16, gsrc + (size_t)ch * 16);
  }
  wait_async0();
  __syncthreads();

  int wave = tid >> 5, lane = tid & 31, h = lane >> 4, q = lane & 15;
  for (int i = 0; i < 2; ++i) {
    int ko0 = (wave * 2 + i) << 4;
    v8f acc[4];
#pragma unroll
    for (int j = 0; j < 4; ++j) acc[j] = (v8f){0.f, 0.f, 0.f, 0.f, 0.f, 0.f, 0.f, 0.f};
    for (int t = 0; t < 3; ++t)
      gemm256x4(w3 + (size_t)t * K_DIM * K_DIM + (ko0 + q) * K_DIM,
                tA + (q + t) * K_DIM, h, acc);
    gemm256x4(wsc + (size_t)(ko0 + q) * K_DIM, tY + q * K_DIM, h, acc);
    float bs[8];
#pragma unroll
    for (int r = 0; r < 8; ++r) bs[r] = b3[ko0 + 8 * h + r] + bsc[ko0 + 8 * h + r];
#pragma unroll
    for (int j = 0; j < 4; ++j) {
      float v[8];
#pragma unroll
      for (int r = 0; r < 8; ++r) v[r] = fmaxf(acc[j][r] + bs[r], 0.f);
      uint4 pk;
      pk.x = pk2(v[0], v[1]); pk.y = pk2(v[2], v[3]);
      pk.z = pk2(v[4], v[5]); pk.w = pk2(v[6], v[7]);
      *(uint4*)(out + ((size_t)n * L_DIM + l0 + 16 * j + q) * K_DIM + ko0 + 8 * h) = pk;
    }
  }
}

// ---------------------------------------------------------------------------
// Reference's (B*C,K,L)->(B,C,L,K) reshape is a reinterpretation: its "mean
// over L" at output k2 averages yfinal over {(k,l): l % 256 == k2}. With
// act[n][l][k] this is 8 rows (l = j*256+k2) fully summed over k.
// z = mean + min_L(d); a = gelu(z). One wave per (n,k2).
// ---------------------------------------------------------------------------
__global__ void __launch_bounds__(256) k_reduce(
    const unsigned short* __restrict__ yf, const float* __restrict__ dmin,
    float* __restrict__ z, float* __restrict__ a) {
  int g = blockIdx.x * 8 + (threadIdx.x >> 5);
  int lane = threadIdx.x & 31;
  int n = g >> 8, k2 = g & 255;
  const unsigned short* p = yf + (size_t)n * L_DIM * K_DIM;
  float s = 0.f;
  for (int e = lane; e < 2048; e += 32) {
    int j = e >> 8, k = e & 255;                 // lane-contiguous in k
    s += bf2f(p[(size_t)(j * 256 + k2) * K_DIM + k]);
  }
  for (int off = 16; off; off >>= 1) s += __shfl_xor(s, off, 32);
  if (lane == 0) {
    float zz = s * (1.f / 2048.f) + dmin[n * K_DIM + k2];
    z[n * K_DIM + k2] = zz;
    a[n * K_DIM + k2] = gelu_fast(zz);
  }
}

// ---------------------------------------------------------------------------
// Tiny attention over C=3 + output proj + LayerNorm + channel mean.
// One block per batch b, 256 threads. ~40 MFLOP total -> scalar VALU is fine.
// ---------------------------------------------------------------------------
__global__ void __launch_bounds__(256) k_attn(
    const float* __restrict__ a, const float* __restrict__ z,
    const float* __restrict__ wq, const float* __restrict__ bq,
    const float* __restrict__ wk, const float* __restrict__ bk,
    const float* __restrict__ wv, const float* __restrict__ bv,
    const float* __restrict__ wo, const float* __restrict__ bo,
    const float* __restrict__ ln_g, const float* __restrict__ ln_b,
    float* __restrict__ out) {
  __shared__ float aS[3 * 256], qS[3 * 256], kS[3 * 256], vS[3 * 256];
  __shared__ float cS[3 * 256], oS[3 * 256];
  __shared__ float scr[18];
  __shared__ float muS[3], varS[3];
  int b = blockIdx.x, tid = threadIdx.x;

  for (int c = 0; c < 3; ++c) aS[c * 256 + tid] = a[((size_t)b * 3 + c) * 256 + tid];
  __syncthreads();

  for (int c = 0; c < 3; ++c) {
    float q_ = bq[tid], k_ = bk[tid], v_ = bv[tid];
    for (int j = 0; j < 256; ++j) {
      float av = aS[c * 256 + j];
      q_ += av * wq[j * 256 + tid];
      k_ += av * wk[j * 256 + tid];
      v_ += av * wv[j * 256 + tid];
    }
    qS[c * 256 + tid] = q_; kS[c * 256 + tid] = k_; vS[c * 256 + tid] = v_;
  }
  __syncthreads();

  if (tid < 18) {                                 // 2 heads x 3 q x 3 k
    int hh = tid / 9, cq = (tid % 9) / 3, ck = tid % 3;
    float s = 0.f;
    for (int d = 0; d < DK_DIM; ++d)
      s += qS[cq * 256 + hh * 128 + d] * kS[ck * 256 + hh * 128 + d];
    scr[tid] = s * 0.088388347648318447f;         // 1/sqrt(128)
  }
  __syncthreads();
  if (tid < 6) {                                  // softmax over 3 keys
    int base = (tid / 3) * 9 + (tid % 3) * 3;
    float m = fmaxf(scr[base], fmaxf(scr[base + 1], scr[base + 2]));
    float e0 = __expf(scr[base] - m), e1 = __expf(scr[base + 1] - m), e2 = __expf(scr[base + 2] - m);
    float inv = 1.f / (e0 + e1 + e2);
    scr[base] = e0 * inv; scr[base + 1] = e1 * inv; scr[base + 2] = e2 * inv;
  }
  __syncthreads();
  {
    int hh = tid >> 7;                            // tid = hh*128 + d
    for (int cq = 0; cq < 3; ++cq) {
      float s = 0.f;
      for (int ck = 0; ck < 3; ++ck)
        s += scr[hh * 9 + cq * 3 + ck] * vS[ck * 256 + tid];
      cS[cq * 256 + tid] = s;
    }
  }
  __syncthreads();
  for (int c = 0; c < 3; ++c) {
    float acc = bo[tid] + z[((size_t)b * 3 + c) * 256 + tid];
    for (int j = 0; j < 256; ++j) acc += cS[c * 256 + j] * wo[j * 256 + tid];
    oS[c * 256 + tid] = acc;
  }
  __syncthreads();
  int wave = tid >> 5, lane = tid & 31;
  if (wave < 3) {
    float s = 0.f, ss = 0.f;
    for (int j = lane; j < 256; j += 32) { float v = oS[wave * 256 + j]; s += v; ss += v * v; }
    for (int off = 16; off; off >>= 1) { s += __shfl_xor(s, off, 32); ss += __shfl_xor(ss, off, 32); }
    if (lane == 0) { float m = s / 256.f; muS[wave] = m; varS[wave] = ss / 256.f - m * m; }
  }
  __syncthreads();
  float o = 0.f;
  for (int c = 0; c < 3; ++c)
    o += (oS[c * 256 + tid] - muS[c]) * rsqrtf(varS[c] + 1e-5f) * ln_g[tid] + ln_b[tid];
  out[(size_t)b * 256 + tid] = o * (1.f / 3.f);
}

// ---------------------------------------------------------------------------
extern "C" void kernel_launch(void* const* d_in, const int* in_sizes, int n_in,
                              void* d_out, int out_size, void* d_ws, size_t ws_size,
                              hipStream_t stream) {
  (void)in_sizes; (void)n_in; (void)out_size; (void)ws_size;
  const float* x    = (const float*)d_in[0];
  const float* shp  = (const float*)d_in[1];
  const float* w1   = (const float*)d_in[2];
  const float* b1   = (const float*)d_in[3];
  const float* w2   = (const float*)d_in[4];
  const float* b2   = (const float*)d_in[5];
  const float* w3   = (const float*)d_in[6];
  const float* b3   = (const float*)d_in[7];
  const float* wsc  = (const float*)d_in[8];
  const float* bsc  = (const float*)d_in[9];
  const float* wq   = (const float*)d_in[10];
  const float* bq   = (const float*)d_in[11];
  const float* wk   = (const float*)d_in[12];
  const float* bk   = (const float*)d_in[13];
  const float* wv   = (const float*)d_in[14];
  const float* bv   = (const float*)d_in[15];
  const float* wo   = (const float*)d_in[16];
  const float* bo   = (const float*)d_in[17];
  const float* ln_g = (const float*)d_in[18];
  const float* ln_b = (const float*)d_in[19];

  char* ws = (char*)d_ws;
  size_t off = 0;
  auto alloc = [&](size_t bytes) -> void* {
    void* p = (void*)(ws + off);
    off += (bytes + 255) & ~(size_t)255;
    return p;
  };
  const size_t ACT = (size_t)NS * K_DIM * L_DIM * sizeof(unsigned short); // ~100.7 MB
  unsigned short* ybf   = (unsigned short*)alloc(ACT);
  unsigned short* h1bf  = (unsigned short*)alloc(ACT);
  unsigned short* h2bf  = (unsigned short*)alloc(ACT);
  unsigned short* w1bf  = (unsigned short*)alloc((size_t)7 * K_DIM * K_DIM * 2);
  unsigned short* w2bf  = (unsigned short*)alloc((size_t)5 * K_DIM * K_DIM * 2);
  unsigned short* w3bf  = (unsigned short*)alloc((size_t)3 * K_DIM * K_DIM * 2);
  unsigned short* wscbf = (unsigned short*)alloc((size_t)1 * K_DIM * K_DIM * 2);
  unsigned short* shpbf = (unsigned short*)alloc((size_t)C_DIM * K_DIM * S_DIM * 2);
  float* x2buf = (float*)alloc((size_t)NS * L_DIM * 4);
  float* s2buf = (float*)alloc((size_t)C_DIM * K_DIM * 4);
  float* dminb = (float*)alloc((size_t)NS * K_DIM * 4);
  float* zbuf  = (float*)alloc((size_t)NS * K_DIM * 4);
  float* abuf  = (float*)alloc((size_t)NS * K_DIM * 4);

  k_init<<<(NS * K_DIM + 255) / 256, 256, 0, stream>>>(shp, s2buf, dminb);
  k_scvt<<<(C_DIM * K_DIM * S_DIM + 255) / 256, 256, 0, stream>>>(shp, shpbf);
  k_x2<<<(NS * L_DIM + 255) / 256, 256, 0, stream>>>(x, x2buf);
  k_wcvt<<<(7 * K_DIM * K_DIM + 255) / 256, 256, 0, stream>>>(w1, w1bf, 7);
  k_wcvt<<<(5 * K_DIM * K_DIM + 255) / 256, 256, 0, stream>>>(w2, w2bf, 5);
  k_wcvt<<<(3 * K_DIM * K_DIM + 255) / 256, 256, 0, stream>>>(w3, w3bf, 3);
  k_wcvt<<<(1 * K_DIM * K_DIM + 255) / 256, 256, 0, stream>>>(wsc, wscbf, 1);

  k_dist<<<NS * 128, 256, 0, stream>>>(x, shpbf, x2buf, s2buf, ybf, dminb);

  k_conv<7, true><<<NS * 32, 256, 0, stream>>>(ybf, w1bf, b1, h1bf);
  k_conv<5, true><<<NS * 32, 256, 0, stream>>>(h1bf, w2bf, b2, h2bf);
  k_conv3skip<<<NS * 32, 256, 0, stream>>>(h2bf, ybf, w3bf, wscbf, b3, bsc, h1bf);

  k_reduce<<<NS * K_DIM / 8, 256, 0, stream>>>(h1bf, dminb, zbuf, abuf);
  k_attn<<<B_DIM, 256, 0, stream>>>(abuf, zbuf, wq, bq, wk, bk, wv, bv,
                                    wo, bo, ln_g, ln_b, (float*)d_out);
}
